// MoELayer_82308753260799
// MI455X (gfx1250) — matrix operand order repaired and checked by reading, not compiled, not run on hardware
//
#include <hip/hip_runtime.h>
#include <hip/hip_bf16.h>

typedef _Float16 half16 __attribute__((ext_vector_type(16)));
typedef float    float8 __attribute__((ext_vector_type(8)));

union F16Frag { half16 v; uint4 u[2]; };

#define T_TOK   32768
#define DIM     1024
#define HID     4096
#define NEXP    4
#define CAPACITY 10240
#define NSEG    512          /* 64 tokens per segment */

#define ASYNC_WAIT0() asm volatile("s_wait_asynccnt 0x0" ::: "memory")

__device__ __forceinline__ float gelu_exact(float x) {
    return 0.5f * x * (1.0f + erff(x * 0.70710678118654752440f));
}

// ---------------------------------------------------------------- zero fill
__global__ void zero_kernel(float* p, size_t n) {
    size_t i = (size_t)blockIdx.x * blockDim.x + threadIdx.x;
    size_t stride = (size_t)gridDim.x * blockDim.x;
    for (; i < n; i += stride) p[i] = 0.0f;
}

// ------------------------------------------- f32 [R,C] -> f16 [C,R] per expert
__global__ void transpose_convert_kernel(const float* __restrict__ src,
                                         _Float16* __restrict__ dst,
                                         int R, int C) {
    __shared__ float tile[32][33];
    const float* s = src + (size_t)blockIdx.z * R * C;
    _Float16*    d = dst + (size_t)blockIdx.z * R * C;
    int tx = threadIdx.x, ty = threadIdx.y;
    int c0 = blockIdx.x * 32, r0 = blockIdx.y * 32;
#pragma unroll
    for (int i = 0; i < 4; ++i)
        tile[ty + i * 8][tx] = s[(size_t)(r0 + ty + i * 8) * C + (c0 + tx)];
    __syncthreads();
#pragma unroll
    for (int i = 0; i < 4; ++i)
        d[(size_t)(c0 + ty + i * 8) * R + (r0 + tx)] =
            (_Float16)tile[tx][ty + i * 8];
}

// ---------------------------------------------------------------- router
// block = 256 threads = 8 waves, wave handles 8 tokens -> 64 tokens / block
__global__ __launch_bounds__(256) void router_kernel(
    const float* __restrict__ x, const float* __restrict__ Wg,
    const float* __restrict__ bg, unsigned char* __restrict__ mask,
    int* __restrict__ segcnt, int* __restrict__ counts_loss) {
    __shared__ int scnt[NEXP];
    int tid = threadIdx.x;
    if (tid < NEXP) scnt[tid] = 0;
    __syncthreads();
    int wv = tid >> 5, lane = tid & 31;
#pragma unroll 1
    for (int it = 0; it < 8; ++it) {
        int t = blockIdx.x * 64 + wv * 8 + it;
        const float* xr = x + (size_t)t * DIM;
        float s0 = 0.f, s1 = 0.f, s2 = 0.f, s3 = 0.f;
#pragma unroll 4
        for (int j = 0; j < 32; ++j) {
            int d = j * 32 + lane;
            float xv = xr[d];
            const float* wr = Wg + (size_t)d * NEXP;
            s0 += xv * wr[0]; s1 += xv * wr[1];
            s2 += xv * wr[2]; s3 += xv * wr[3];
        }
#pragma unroll
        for (int off = 16; off > 0; off >>= 1) {
            s0 += __shfl_xor(s0, off, 32);
            s1 += __shfl_xor(s1, off, 32);
            s2 += __shfl_xor(s2, off, 32);
            s3 += __shfl_xor(s3, off, 32);
        }
        if (lane == 0) {
            float v[NEXP] = { s0 + bg[0], s1 + bg[1], s2 + bg[2], s3 + bg[3] };
            int i0 = 0;                       // jax top_k: ties -> lowest index
            for (int e = 1; e < NEXP; ++e) if (v[e] > v[i0]) i0 = e;
            int i1 = -1;
            for (int e = 0; e < NEXP; ++e) {
                if (e == i0) continue;
                if (i1 < 0 || v[e] > v[i1]) i1 = e;
            }
            mask[t] = (unsigned char)((1u << i0) | (1u << i1));
            atomicAdd(&scnt[i0], 1); atomicAdd(&scnt[i1], 1);
            atomicAdd(&counts_loss[i0], 1); atomicAdd(&counts_loss[i1], 1);
        }
    }
    __syncthreads();
    if (tid < NEXP) segcnt[tid * NSEG + blockIdx.x] = scnt[tid];
}

// ------------------------------------ exclusive scan over segments + loss
__global__ void scan_kernel(const int* __restrict__ segcnt,
                            int* __restrict__ segoff, int* __restrict__ cnt,
                            const int* __restrict__ counts_loss,
                            float* __restrict__ loss_out) {
    int e = threadIdx.x;
    if (e < NEXP) {
        int run = 0;
        for (int s = 0; s < NSEG; ++s) {
            segoff[e * NSEG + s] = run;
            run += segcnt[e * NSEG + s];
        }
        cnt[e] = run < CAPACITY ? run : CAPACITY;
    }
    if (e == NEXP) {
        float c[NEXP], s = 0.f;
        for (int i = 0; i < NEXP; ++i) { c[i] = (float)counts_loss[i]; s += c[i]; }
        float m = s / (float)NEXP, var = 0.f;
        for (int i = 0; i < NEXP; ++i) var += (c[i] - m) * (c[i] - m);
        var /= (float)(NEXP - 1);           // ddof=1 (torch unbiased)
        loss_out[0] = sqrtf(var) / m;
    }
}

// ---------------------------------------- FIFO assignment within capacity
__global__ void assign_kernel(const unsigned char* __restrict__ mask,
                              const int* __restrict__ segoff,
                              int* __restrict__ toklist) {
    int e = threadIdx.x;
    if (e >= NEXP) return;
    int blk = blockIdx.x;
    int pos = segoff[e * NSEG + blk];
    int t0 = blk * 64;
    for (int i = 0; i < 64; ++i) {
        if ((mask[t0 + i] >> e) & 1) {
            if (pos < CAPACITY) toklist[e * CAPACITY + pos] = t0 + i;
            ++pos;
        }
    }
}

// --------------------------------------- gather x rows -> f16 Xe[cap, D]
__global__ __launch_bounds__(256) void gather_kernel(
    const float* __restrict__ x, const int* __restrict__ toklist_e,
    const int* __restrict__ cnt_e, _Float16* __restrict__ Xe) {
    int r = blockIdx.x * 8 + (threadIdx.x >> 5);
    int lane = threadIdx.x & 31;
    int cnt = *cnt_e;
    union { _Float16 h[4]; uint2 u; } pk;
    _Float16* dst = Xe + (size_t)r * DIM;
    if (r < cnt) {
        int tok = toklist_e[r];
        const float* src = x + (size_t)tok * DIM;
#pragma unroll
        for (int j = 0; j < 8; ++j) {
            int idx = j * 128 + lane * 4;
            float4 v = *(const float4*)(src + idx);
            pk.h[0] = (_Float16)v.x; pk.h[1] = (_Float16)v.y;
            pk.h[2] = (_Float16)v.z; pk.h[3] = (_Float16)v.w;
            *(uint2*)(dst + idx) = pk.u;
        }
    } else {
        uint2 z; z.x = 0u; z.y = 0u;
#pragma unroll
        for (int j = 0; j < 8; ++j)
            *(uint2*)(dst + j * 128 + lane * 4) = z;
    }
}

// --------------------------------------------------- tiled WMMA GEMM
// C[M,N] = A[M,K] * BT[N,K]^T + bias   (A,BT f16 row-major, acc f32)
// Double-buffered LDS tiles filled by GLOBAL_LOAD_ASYNC_TO_LDS_B128 (ASYNCcnt),
// overlapping the DMA of tile k+1 with WMMA on tile k.
// EPI==0 : out = gelu(C) -> f16 Hout[M,N]
// EPI==1 : Out[toklist[m]*N + n] += C  (only rows m < *cnt_e)
template <int EPI>
__global__ __launch_bounds__(256) void wmma_gemm_kernel(
    const _Float16* __restrict__ A, const _Float16* __restrict__ BT,
    const float* __restrict__ bias, _Float16* __restrict__ Hout,
    float* __restrict__ Out, const int* __restrict__ toklist_e,
    const int* __restrict__ cnt_e, int M, int N, int K) {
    // +8-half pad: row pitch 144B (keeps 16B alignment, kills bank conflicts)
    __shared__ __align__(128) _Float16 As[2][128][72];
    __shared__ __align__(128) _Float16 Bs[2][128][72];

    int tid = threadIdx.x;
    int lane = tid & 31, wv = tid >> 5;
    int wm = wv >> 2;            // 0..1  (64 rows each)
    int wn = wv & 3;             // 0..3  (32 cols each)
    int lhi = lane >> 4, llo = lane & 15;
    int m0 = blockIdx.y * 128, n0 = blockIdx.x * 128;

    // async-copy assignments: thread -> (row = tid/8 + p*32, 16B chunk = tid%8)
    int ldr = tid >> 3;
    unsigned ldcB = (unsigned)(tid & 7) * 16u;            // byte col in tile
    unsigned aLds[2], bLds[2];
#pragma unroll
    for (int b = 0; b < 2; ++b) {
        aLds[b] = (unsigned)(uintptr_t)(&As[b][ldr][0]) + ldcB;
        bLds[b] = (unsigned)(uintptr_t)(&Bs[b][ldr][0]) + ldcB;
    }

    auto issue_tile = [&](int kt, int b) {
        // byte offsets into A / BT (GVS mode: SGPR base + 32-bit VGPR offset)
        unsigned abyte =
            (unsigned)(((size_t)(m0 + ldr) * K + (size_t)kt * 64) * 2) + ldcB;
        unsigned bbyte =
            (unsigned)(((size_t)(n0 + ldr) * K + (size_t)kt * 64) * 2) + ldcB;
        unsigned stride = (unsigned)K * 64u;  // 32 rows * K halves * 2B
#pragma unroll
        for (int p = 0; p < 4; ++p) {
            asm volatile("global_load_async_to_lds_b128 %0, %1, %2"
                         :: "v"(aLds[b] + (unsigned)p * 32u * 144u),
                            "v"(abyte + (unsigned)p * stride),
                            "s"(A)
                         : "memory");
            asm volatile("global_load_async_to_lds_b128 %0, %1, %2"
                         :: "v"(bLds[b] + (unsigned)p * 32u * 144u),
                            "v"(bbyte + (unsigned)p * stride),
                            "s"(BT)
                         : "memory");
        }
    };

    const float8 z8 = {0.f, 0.f, 0.f, 0.f, 0.f, 0.f, 0.f, 0.f};
    float8 acc[4][2];
#pragma unroll
    for (int i = 0; i < 4; ++i)
#pragma unroll
        for (int j = 0; j < 2; ++j) acc[i][j] = z8;

    int KT = K >> 6;             // K / 64
    issue_tile(0, 0);
    ASYNC_WAIT0();
    __syncthreads();

    for (int kt = 0; kt < KT; ++kt) {
        int cur = kt & 1;
        if (kt + 1 < KT) issue_tile(kt + 1, cur ^ 1);   // DMA next tile

        // compute on current buffer: 2 x (12 ds_load_b128 -> 8 wmma)
#pragma unroll
        for (int ks = 0; ks < 2; ++ks) {
            F16Frag fa[4], fb[2];
#pragma unroll
            for (int j = 0; j < 2; ++j) {
                // B frag: lane = column N, K split by lane half (ISA 7.12.2)
                const _Float16* bp =
                    &Bs[cur][wn * 32 + j * 16 + llo][ks * 32 + lhi * 16];
                fb[j].u[0] = *(const uint4*)bp;
                fb[j].u[1] = *(const uint4*)(bp + 8);
            }
#pragma unroll
            for (int i = 0; i < 4; ++i) {
                // A frag: lane = row M, per-lane K = {kb+16*lhi..+15}
                const _Float16* ap =
                    &As[cur][wm * 64 + i * 16 + llo][ks * 32 + lhi * 16];
                fa[i].u[0] = *(const uint4*)ap;
                fa[i].u[1] = *(const uint4*)(ap + 8);
            }
#pragma unroll
            for (int i = 0; i < 4; ++i)
#pragma unroll
                for (int j = 0; j < 2; ++j)
                    acc[i][j] = __builtin_amdgcn_wmma_f32_16x16x32_f16(
                        false, fa[i].v, false, fb[j].v, (short)0, acc[i][j],
                        false, false);
        }

        if (kt + 1 < KT) {
            ASYNC_WAIT0();       // next tile landed in LDS
            __syncthreads();     // all waves done reading cur + writes visible
        }
    }

    int cnt = 0;
    if (EPI == 1) cnt = *cnt_e;
#pragma unroll
    for (int i = 0; i < 4; ++i) {
#pragma unroll
        for (int j = 0; j < 2; ++j) {
            int mbase = m0 + wm * 64 + i * 16 + lhi * 8;
            int n = n0 + wn * 32 + j * 16 + llo;
            float b = bias[n];
#pragma unroll
            for (int r = 0; r < 8; ++r) {
                float v = acc[i][j][r] + b;
                int m = mbase + r;
                if (EPI == 0) {
                    Hout[(size_t)m * N + n] = (_Float16)gelu_exact(v);
                } else {
                    if (m < cnt) {
                        int tok = toklist_e[m];
                        Out[(size_t)tok * N + n] += v;   // rows unique per expert
                    }
                }
            }
        }
    }
}

// ---------------------------------------------------------------- launch
extern "C" void kernel_launch(void* const* d_in, const int* in_sizes, int n_in,
                              void* d_out, int out_size, void* d_ws,
                              size_t ws_size, hipStream_t stream) {
    const float* x  = (const float*)d_in[0];   // [T, D]
    const float* Wg = (const float*)d_in[1];   // [D, E]
    const float* bg = (const float*)d_in[2];   // [E]
    const float* W1 = (const float*)d_in[3];   // [E, D, H]
    const float* b1 = (const float*)d_in[4];   // [E, H]
    const float* W2 = (const float*)d_in[5];   // [E, H, D]
    const float* b2 = (const float*)d_in[6];   // [E, D]
    float* out = (float*)d_out;                // [T*D] output ++ [1] loss

    // workspace carve (256B aligned)
    char* base = (char*)d_ws;
    size_t off = 0;
    auto carve = [&](size_t bytes) -> char* {
        off = (off + 255) & ~(size_t)255;
        char* p = base + off;
        off += bytes;
        return p;
    };
    unsigned char* mask = (unsigned char*)carve(T_TOK);
    int* segcnt      = (int*)carve(sizeof(int) * NEXP * NSEG);
    int* segoff      = (int*)carve(sizeof(int) * NEXP * NSEG);
    int* counts_loss = (int*)carve(sizeof(int) * NEXP);
    int* cnt         = (int*)carve(sizeof(int) * NEXP);
    int* toklist     = (int*)carve(sizeof(int) * NEXP * CAPACITY);
    _Float16* W1T = (_Float16*)carve((size_t)NEXP * DIM * HID * 2); // [E,H,D]
    _Float16* W2T = (_Float16*)carve((size_t)NEXP * HID * DIM * 2); // [E,D,H]
    _Float16* Xe  = (_Float16*)carve((size_t)CAPACITY * DIM * 2);   // [cap,D]
    _Float16* H1  = (_Float16*)carve((size_t)CAPACITY * HID * 2);   // [cap,H]
    (void)in_sizes; (void)n_in; (void)out_size; (void)ws_size;

    // 1. zero output (+loss slot) and loss counters
    zero_kernel<<<2048, 256, 0, stream>>>(out, (size_t)T_TOK * DIM + 1);
    zero_kernel<<<1, 32, 0, stream>>>((float*)counts_loss, NEXP);

    // 2. weight transpose-convert: W1 [E,D,H] -> W1T [E,H,D]; W2 [E,H,D] -> W2T [E,D,H]
    {
        dim3 blk(32, 8);
        transpose_convert_kernel<<<dim3(HID / 32, DIM / 32, NEXP), blk, 0,
                                   stream>>>(W1, W1T, DIM, HID);
        transpose_convert_kernel<<<dim3(DIM / 32, HID / 32, NEXP), blk, 0,
                                   stream>>>(W2, W2T, HID, DIM);
    }

    // 3. router + segment counts
    router_kernel<<<NSEG, 256, 0, stream>>>(x, Wg, bg, mask, segcnt,
                                            counts_loss);
    // 4. ordered scan (FIFO capacity) + loss
    scan_kernel<<<1, 32, 0, stream>>>(segcnt, segoff, cnt, counts_loss,
                                      out + (size_t)T_TOK * DIM);
    // 5. token assignment
    assign_kernel<<<NSEG, 32, 0, stream>>>(mask, segoff, toklist);

    // 6. experts (sequential on stream -> scatter += is race-free)
    for (int e = 0; e < NEXP; ++e) {
        gather_kernel<<<CAPACITY / 8, 256, 0, stream>>>(
            x, toklist + (size_t)e * CAPACITY, cnt + e, Xe);
        wmma_gemm_kernel<0><<<dim3(HID / 128, CAPACITY / 128), 256, 0,
                              stream>>>(
            Xe, W1T + (size_t)e * HID * DIM, b1 + (size_t)e * HID, H1, nullptr,
            nullptr, nullptr, CAPACITY, HID, DIM);
        wmma_gemm_kernel<1><<<dim3(DIM / 128, CAPACITY / 128), 256, 0,
                              stream>>>(
            H1, W2T + (size_t)e * DIM * HID, b2 + (size_t)e * DIM, nullptr, out,
            toklist + (size_t)e * CAPACITY, cnt + e, CAPACITY, DIM, HID);
    }
}
